// SelfAttention_6571299963010
// MI455X (gfx1250) — matrix operand order repaired and checked by reading, not compiled, not run on hardware
//
#include <hip/hip_runtime.h>

// ---------------------------------------------------------------------------
// SelfAttention (reference's degenerate einsum) on MI455X / gfx1250.
//
// Identity: softmax rows sum to 1 and attn is contracted over BOTH q and k,
// so  out = 2048 * (x @ Wv + bv) @ Wo + bo.  Two 4096x1024x1024 GEMMs
// (17.2 GFLOP); Wq/bq/Wk/bk are provably unused.
//
// GEMM: bf16 WMMA (v_wmma_f32_16x16x32_bf16, fp32 accum), 128x256 block tile,
// 8 wave32 each owning 64x64 (16 accumulators). Global->LDS staging uses the
// CDNA5 async DMA path (global_load_async_to_lds_b128 + s_wait_asynccnt),
// double-buffered in LDS (48 KB of the 320 KB WGP pool).
// ---------------------------------------------------------------------------

typedef __attribute__((ext_vector_type(16))) __bf16 v16bf;
typedef __attribute__((ext_vector_type(8)))  float  v8f;
typedef __attribute__((ext_vector_type(4)))  int    v4i;
typedef unsigned short u16;

// Pointer-to-v4i in specific address spaces (AS1 = global, AS3 = LDS).
typedef __attribute__((address_space(1))) v4i*  as1_v4i;
typedef __attribute__((address_space(3))) v4i*  as3_v4i;
typedef __attribute__((address_space(1))) void* as1_void;
typedef __attribute__((address_space(3))) void* as3_void;

union FragU { v16bf v; uint4 q[2]; };

__device__ __forceinline__ u16 f2bf(float f) {
  unsigned u = __builtin_bit_cast(unsigned, f);
  u += 0x7FFFu + ((u >> 16) & 1u);          // round-to-nearest-even
  return (u16)(u >> 16);
}

// ---- CDNA5 async global->LDS copy (ASYNCcnt path) -------------------------
__device__ __forceinline__ void async_b128(const u16* g, u16* l) {
#if __has_builtin(__builtin_amdgcn_global_load_async_to_lds_b128)
  // Prototype (from hipcc diagnostic): (v4i addrspace(1)*, v4i addrspace(3)*,
  // imm offset, imm cpol). Cast addrspace first, then pointee type.
  __builtin_amdgcn_global_load_async_to_lds_b128(
      (as1_v4i)(as1_void)(void*)const_cast<u16*>(g),
      (as3_v4i)(as3_void)(void*)l,
      0, 0);
#else
  unsigned lo = (unsigned)(unsigned long long)(as3_void)(void*)l;
  asm volatile("global_load_async_to_lds_b128 %0, %1, off"
               :: "v"(lo), "v"(g) : "memory");
#endif
}

__device__ __forceinline__ void wait_async0() {
#if __has_builtin(__builtin_amdgcn_s_wait_asynccnt)
  __builtin_amdgcn_s_wait_asynccnt(0);
#else
  asm volatile("s_wait_asynccnt 0x0" ::: "memory");
#endif
}

// ---------------- fp32 -> bf16 (same layout), vectorized -------------------
__global__ __launch_bounds__(256)
void f32_to_bf16_vec(const float* __restrict__ in, u16* __restrict__ out, int n4) {
  int i = blockIdx.x * blockDim.x + threadIdx.x;
  if (i < n4) {
    float4 f = ((const float4*)in)[i];
    ushort4 o;
    o.x = f2bf(f.x); o.y = f2bf(f.y); o.z = f2bf(f.z); o.w = f2bf(f.w);
    ((ushort4*)out)[i] = o;
  }
}

// -------- fp32 [K][N] -> bf16 [N][K] (transpose) via LDS tile --------------
__global__ __launch_bounds__(256)
void transpose_f32_to_bf16(const float* __restrict__ W, u16* __restrict__ Wt,
                           int K, int N) {
  __shared__ float tile[32][33];
  const int k0 = blockIdx.y * 32, n0 = blockIdx.x * 32;
  const int tx = threadIdx.x, ty = threadIdx.y;   // blockDim = (32,8)
#pragma unroll
  for (int j = 0; j < 4; ++j)
    tile[ty + j * 8][tx] = W[(size_t)(k0 + ty + j * 8) * N + (n0 + tx)];
  __syncthreads();
#pragma unroll
  for (int j = 0; j < 4; ++j)
    Wt[(size_t)(n0 + ty + j * 8) * K + (k0 + tx)] = f2bf(tile[tx][ty + j * 8]);
}

// ---------------------------------------------------------------------------
// Tiled bf16 WMMA GEMM:  C = alpha * (A @ B) + bias,  B given transposed.
//   A  : [M][K] bf16 row-major      Bt : [N][K] bf16 row-major
//   C  : [M][N] bf16 (OUT_BF16) or fp32
// Block: 256 threads (8 wave32), tile BM x BN = 128 x 256, K-step 32.
// Wave grid 2x4 -> each wave owns 64x64 = 4x4 WMMA accumulators.
// ---------------------------------------------------------------------------
#define BM 128
#define BN 256
#define BK 32

template <bool OUT_BF16>
__global__ __launch_bounds__(256)
void gemm_bf16_wmma(const u16* __restrict__ A, const u16* __restrict__ Bt,
                    const float* __restrict__ bias, float alpha,
                    void* __restrict__ Cout, int M, int N, int K) {
  __shared__ u16 As[2][BM * BK];   // 2 x 8 KB
  __shared__ u16 Bs[2][BN * BK];   // 2 x 16 KB

  const int tid  = threadIdx.x;
  const int lane = tid & 31;
  const int wid  = tid >> 5;           // 0..7
  const int wm   = (wid >> 2) * 64;    // wave row base in block tile (0,64)
  const int wn   = (wid & 3) * 64;     // wave col base in block tile (0..192)

  const int m0 = blockIdx.y * BM;
  const int n0 = blockIdx.x * BN;
  const int NK = K / BK;

  // Async staging: tiles are rows of 64B (BK bf16) = 4 x 16B chunks per row.
  // A: 512 chunks (2/thread), B: 1024 chunks (4/thread).
  auto stage = [&](int kt, int buf) {
    const u16* ap = A  + (size_t)m0 * K + (size_t)kt * BK;
    const u16* bp = Bt + (size_t)n0 * K + (size_t)kt * BK;
#pragma unroll
    for (int i = 0; i < 2; ++i) {
      const int c = tid + i * 256, row = c >> 2, seg = (c & 3) * 8;
      async_b128(ap + (size_t)row * K + seg, &As[buf][row * BK + seg]);
    }
#pragma unroll
    for (int i = 0; i < 4; ++i) {
      const int c = tid + i * 256, row = c >> 2, seg = (c & 3) * 8;
      async_b128(bp + (size_t)row * K + seg, &Bs[buf][row * BK + seg]);
    }
  };

  const v8f zero = {0.f, 0.f, 0.f, 0.f, 0.f, 0.f, 0.f, 0.f};
  v8f acc[4][4];
#pragma unroll
  for (int i = 0; i < 4; ++i)
#pragma unroll
    for (int j = 0; j < 4; ++j) acc[i][j] = zero;

  stage(0, 0);
  wait_async0();
  __syncthreads();

  int buf = 0;
  for (int kt = 0; kt < NK; ++kt) {
    if (kt + 1 < NK) stage(kt + 1, buf ^ 1);     // DMA overlaps the WMMAs below
    if (kt + 2 < NK) {                           // L2 prefetch two tiles ahead
      __builtin_prefetch(A  + (size_t)(m0 + (tid >> 2)) * K + (size_t)(kt + 2) * BK, 0, 1);
      __builtin_prefetch(Bt + (size_t)(n0 + (tid >> 2)) * K + (size_t)(kt + 2) * BK, 0, 1);
    }

    // ---- fragments from LDS, per ISA 16-bit VGPR layouts ----
    // A (16x32): lane<16 -> K 0..7,16..23 ; lane>=16 -> K 8..15,24..31
    FragU af[4], bfrag[4];
    const u16* abase = &As[buf][0];
    const u16* bbase = &Bs[buf][0];
    const int amrow = wm + (lane & 15);
    const int aklo  = (lane >> 4) * 8;
#pragma unroll
    for (int tm = 0; tm < 4; ++tm) {
      const u16* p = abase + (size_t)(amrow + tm * 16) * BK + aklo;
      af[tm].q[0] = *(const uint4*)p;
      af[tm].q[1] = *(const uint4*)(p + 16);
    }
    // B (32x16): lane<16 -> col=lane, K 0..7,16..23 ; lane>=16 -> col=lane-16, K 8..15,24..31
    const int bnrow = wn + (lane & 15);
    const int bklo  = (lane >> 4) * 8;
#pragma unroll
    for (int tn = 0; tn < 4; ++tn) {
      const u16* p = bbase + (size_t)(bnrow + tn * 16) * BK + bklo;
      bfrag[tn].q[0] = *(const uint4*)p;
      bfrag[tn].q[1] = *(const uint4*)(p + 16);
    }

#pragma unroll
    for (int tm = 0; tm < 4; ++tm)
#pragma unroll
      for (int tn = 0; tn < 4; ++tn)
        acc[tm][tn] = __builtin_amdgcn_wmma_f32_16x16x32_bf16(
            false, af[tm].v, false, bfrag[tn].v,
            (short)0, acc[tm][tn], false, false);

    wait_async0();       // our async writes into buf^1 have landed
    __syncthreads();     // everyone's have; reads of buf are long retired
    buf ^= 1;
  }

  // ---- epilogue: C VGPR r holds M=r (lanes 0-15) / M=r+8 (lanes 16-31) ----
  const int crow0 = m0 + wm + (lane >> 4) * 8;
  const int ccol0 = n0 + wn + (lane & 15);
#pragma unroll
  for (int tm = 0; tm < 4; ++tm) {
#pragma unroll
    for (int tn = 0; tn < 4; ++tn) {
      const int col = ccol0 + tn * 16;
      const float bb = bias[col];
#pragma unroll
      for (int r = 0; r < 8; ++r) {
        const int row = crow0 + tm * 16 + r;
        const float val = alpha * acc[tm][tn][r] + bb;
        if constexpr (OUT_BF16)
          ((u16*)Cout)[(size_t)row * N + col] = f2bf(val);
        else
          ((float*)Cout)[(size_t)row * N + col] = val;
      }
    }
  }
}

// ---------------------------------------------------------------------------
extern "C" void kernel_launch(void* const* d_in, const int* in_sizes, int n_in,
                              void* d_out, int out_size, void* d_ws, size_t ws_size,
                              hipStream_t stream) {
  (void)in_sizes; (void)n_in; (void)out_size; (void)ws_size;
  // Inputs: x, Wq, bq, Wk, bk, Wv, bv, Wo, bo  (Wq/bq/Wk/bk unused by identity)
  const float* x  = (const float*)d_in[0];
  const float* Wv = (const float*)d_in[5];
  const float* bv = (const float*)d_in[6];
  const float* Wo = (const float*)d_in[7];
  const float* bo = (const float*)d_in[8];
  float* out = (float*)d_out;

  constexpr int Mrows = 2 * 2048;  // B*S
  constexpr int Kdim  = 1024;
  constexpr int Ndim  = 1024;

  char* ws = (char*)d_ws;
  u16* xb  = (u16*)(ws);                              // 8 MiB  x    bf16 [M][K]
  u16* Wvt = (u16*)(ws + (size_t)8  * 1024 * 1024);   // 2 MiB  Wv^T bf16 [N][K]
  u16* Wot = (u16*)(ws + (size_t)10 * 1024 * 1024);   // 2 MiB  Wo^T bf16 [N][K]
  u16* Vb  = (u16*)(ws + (size_t)12 * 1024 * 1024);   // 8 MiB  V    bf16 [M][N]

  f32_to_bf16_vec<<<(Mrows * Kdim / 4 + 255) / 256, 256, 0, stream>>>(
      x, xb, Mrows * Kdim / 4);

  dim3 tb(32, 8);
  transpose_f32_to_bf16<<<dim3(Ndim / 32, Kdim / 32), tb, 0, stream>>>(Wv, Wvt, Kdim, Ndim);
  transpose_f32_to_bf16<<<dim3(Ndim / 32, Kdim / 32), tb, 0, stream>>>(Wo, Wot, Kdim, Ndim);

  dim3 grid(Ndim / BN, Mrows / BM);
  // V = x @ Wv + bv   (bf16 out, stays hot in the 192 MB L2 for GEMM2)
  gemm_bf16_wmma<true ><<<grid, 256, 0, stream>>>(xb, Wvt, bv, 1.0f,
                                                  (void*)Vb, Mrows, Ndim, Kdim);
  // out = 2048 * (V @ Wo) + bo   (fp32 out)
  gemm_bf16_wmma<false><<<grid, 256, 0, stream>>>(Vb, Wot, bo, 2048.0f,
                                                  (void*)out, Mrows, Ndim, Kdim);
}
